// KLDivLossColBERTInBatch_15152644621119
// MI455X (gfx1250) — matrix operand order repaired and checked by reading, not compiled, not run on hardware
//
#include <hip/hip_runtime.h>
#include <hip/hip_bf16.h>
#include <math.h>

// ---------------------------------------------------------------------------
// Problem dims (fixed by the reference)
// ---------------------------------------------------------------------------
#define BQ   64     // batch
#define LQ   32     // query length
#define LD   256    // doc length
#define DF   128    // feature dim (K)
#define NDOC 2      // docs per query
#define EPSN 1e-12f
#define INBATCH_P 0.5f

typedef __attribute__((ext_vector_type(16))) __bf16 v16bf;
typedef __attribute__((ext_vector_type(8)))  __bf16 v8bf;
typedef __attribute__((ext_vector_type(8)))  float  v8f;

// ---------------------------------------------------------------------------
// Fast max: compiler-visible (so WMMA->VALU hazard NOPs and dual-issue are
// preserved) but with fast-math flags scoped on, so the AMDGPU backend skips
// the IEEE canonicalize self-max.  Inputs are finite WMMA outputs.
// ---------------------------------------------------------------------------
__device__ __forceinline__ float vmax(float a, float b) {
#pragma float_control(precise, off)
  return __builtin_fmaxf(a, b);
}

// ---------------------------------------------------------------------------
// WMMA helper (gfx1250 wave32, V_WMMA_F32_16X16X32_BF16)
// ---------------------------------------------------------------------------
__device__ __forceinline__ v8f wmma_bf16(v16bf a, v16bf b, v8f c) {
  // 8 args: (neg_a, A, neg_b, B, c_mod, C, reuse_a, reuse_b)
  return __builtin_amdgcn_wmma_f32_16x16x32_bf16(false, a, false, b,
                                                 (short)0, c, false, false);
}

// A-fragment, 16x32 bf16, contiguous K layout matching load_fragB (A = doc rows here):
//  lanes 0-15  (M=lane):    elems 0..15 -> K 0..15
//  lanes 16-31 (M=lane-16): elems 0..15 -> K 16..31
// `base` points at (row0, k0) of a row-major [*, DF] matrix.
__device__ __forceinline__ v16bf load_fragA(const __bf16* base, int lane) {
  int m  = lane & 15;
  int hi = (lane >> 4) * 16;
  return *(const v16bf*)(base + m * DF + hi);
}

// B-fragment, 32x16 bf16 (B = q^T; q stored row-major [Lq, DF]):
//  column N = lane&15 (query index c0 + N), lanes 0-15 hold K 0..15,
//  lanes 16-31 hold K 16..31; element e -> K = khi + e.
//  One contiguous 32B load per lane.
__device__ __forceinline__ v16bf load_fragB(const __bf16* qmat, int c0, int k0,
                                            int lane) {
  int nn  = lane & 15;
  int khi = (lane >> 4) * 16;
  return *(const v16bf*)(qmat + (size_t)(c0 + nn) * DF + k0 + khi);
}

// ---------------------------------------------------------------------------
// 1) q: L2-normalize over feature dim -> bf16.  One wave per (b,l) row.
// ---------------------------------------------------------------------------
__global__ __launch_bounds__(256) void norm_q_kernel(const float* __restrict__ q,
                                                     __bf16* __restrict__ Q) {
  int row  = blockIdx.x * 8 + (threadIdx.x >> 5);   // 0 .. B*Lq-1
  int lane = threadIdx.x & 31;
  const float* src = q + (size_t)row * DF;
  float x[4];
  float ss = 0.f;
#pragma unroll
  for (int i = 0; i < 4; ++i) { x[i] = src[lane * 4 + i]; ss += x[i] * x[i]; }
#pragma unroll
  for (int m = 1; m < 32; m <<= 1) ss += __shfl_xor(ss, m, 32);
  float inv = 1.0f / fmaxf(sqrtf(ss), EPSN);
  __bf16* dst = Q + (size_t)row * DF;
#pragma unroll
  for (int i = 0; i < 4; ++i) dst[lane * 4 + i] = (__bf16)(x[i] * inv);
}

// ---------------------------------------------------------------------------
// 2) d: mask, then normalize over the SEQUENCE axis (per feature) -> bf16.
//    Block per (n,b); thread per feature k; reduction strides DF (coalesced).
// ---------------------------------------------------------------------------
__global__ __launch_bounds__(128) void norm_d_kernel(const float* __restrict__ d,
                                                     const int* __restrict__ msk,
                                                     __bf16* __restrict__ Dbf) {
  int nb = blockIdx.x;                               // n*B + b
  const float* src = d + (size_t)nb * LD * DF;
  const int*   mk  = msk + (size_t)nb * LD;
  int k = threadIdx.x;                               // feature 0..127
  float ss = 0.f;
  for (int m = 0; m < LD; ++m) {
    float v = src[(size_t)m * DF + k] * (float)mk[m];
    ss += v * v;
  }
  float inv = 1.0f / fmaxf(sqrtf(ss), EPSN);
  __bf16* dst = Dbf + (size_t)nb * LD * DF;
  for (int m = 0; m < LD; ++m) {
    float v = src[(size_t)m * DF + k] * (float)mk[m];
    dst[(size_t)m * DF + k] = (__bf16)(v * inv);
  }
}

// ---------------------------------------------------------------------------
// 3) MaxSim GEMM, transposed roles: A = doc tile rows (Ld on M), B = q^T
//    (queries on N).  max over Ld is per-lane elementwise v_max of
//    accumulators across the 16 M-tiles (co-executes with WMMA); only 6
//    shuffles per pair at the end.  Each wave independently handles 2 query
//    batches vs one doc -> A-fragments (d) loaded once, shared by 2 pairs.
//    No LDS, no barriers.
// ---------------------------------------------------------------------------
__global__ __launch_bounds__(256) void maxsim_kernel(const __bf16* __restrict__ Q,
                                                     const __bf16* __restrict__ Dbf,
                                                     float* __restrict__ S) {
  int nbd  = blockIdx.x;                 // n*B + bd
  int n    = nbd >> 6;
  int bd   = nbd & 63;
  int wave = threadIdx.x >> 5;
  int lane = threadIdx.x & 31;
  int bq0  = blockIdx.y * 16;
  int bqA  = bq0 + wave;                 // pair 0
  int bqB  = bq0 + 8 + wave;             // pair 1

  const __bf16* dmat = Dbf + (size_t)nbd * LD * DF;
  const __bf16* qmA  = Q + (size_t)bqA * LQ * DF;
  const __bf16* qmB  = Q + (size_t)bqB * LQ * DF;

  // Register-resident q fragments: [pair][ntile][kstep] = 16 x v16bf.
  v16bf qf[2][2][4];
#pragma unroll
  for (int nt = 0; nt < 2; ++nt)
#pragma unroll
    for (int ks = 0; ks < 4; ++ks) {
      qf[0][nt][ks] = load_fragB(qmA, nt * 16, ks * 32, lane);
      qf[1][nt][ks] = load_fragB(qmB, nt * 16, ks * 32, lane);
    }

  // Running max over Ld, elementwise in the C layout. [pair][ntile]
  v8f rmax[2][2];
#pragma unroll
  for (int p = 0; p < 2; ++p)
#pragma unroll
    for (int nt = 0; nt < 2; ++nt)
#pragma unroll
      for (int r = 0; r < 8; ++r) rmax[p][nt][r] = -INFINITY;

  for (int mt = 0; mt < 16; ++mt) {
    const __bf16* abase = dmat + (size_t)mt * 16 * DF;
    if (mt + 1 < 16)  // global_prefetch_b8 of the next doc tile
      __builtin_prefetch(dmat + (size_t)(mt + 1) * 16 * DF, 0, 3);

    v8f acc[2][2] = {};                  // [pair][ntile]
#pragma unroll
    for (int ks = 0; ks < 4; ++ks) {
      v16bf a = load_fragA(abase + ks * 32, lane);   // shared by both pairs
      acc[0][0] = wmma_bf16(a, qf[0][0][ks], acc[0][0]);
      acc[0][1] = wmma_bf16(a, qf[0][1][ks], acc[0][1]);
      acc[1][0] = wmma_bf16(a, qf[1][0][ks], acc[1][0]);
      acc[1][1] = wmma_bf16(a, qf[1][1][ks], acc[1][1]);
    }
#pragma unroll
    for (int p = 0; p < 2; ++p)
#pragma unroll
      for (int nt = 0; nt < 2; ++nt)
#pragma unroll
        for (int r = 0; r < 8; ++r)
          rmax[p][nt][r] = vmax(rmax[p][nt][r], acc[p][nt][r]);
  }

  // Finalize: per pair, reduce 8 in-lane rows, fold lane^16 (rows +8),
  // sum the two query halves, then 4-step lane sum over the 16 query cols.
#pragma unroll
  for (int p = 0; p < 2; ++p) {
    float t[2];
#pragma unroll
    for (int nt = 0; nt < 2; ++nt) {
      float v = vmax(vmax(vmax(rmax[p][nt][0], rmax[p][nt][1]),
                          vmax(rmax[p][nt][2], rmax[p][nt][3])),
                     vmax(vmax(rmax[p][nt][4], rmax[p][nt][5]),
                          vmax(rmax[p][nt][6], rmax[p][nt][7])));
      v = vmax(v, __shfl_xor(v, 16, 32));
      t[nt] = v;
    }
    float s = t[0] + t[1];               // both query halves for col lane&15
    s += __shfl_xor(s, 1, 32);
    s += __shfl_xor(s, 2, 32);
    s += __shfl_xor(s, 4, 32);
    s += __shfl_xor(s, 8, 32);
    int bq = p ? bqB : bqA;
    if (lane == 0)
      S[(size_t)n * BQ * BQ + (size_t)bq * BQ + bd] = s;
  }
}

// ---------------------------------------------------------------------------
// 4) Epilogue: KL(batchmean, log_target) on diagonal scores + 0.5 * in-batch CE
// ---------------------------------------------------------------------------
__global__ __launch_bounds__(64) void loss_kernel(const float* __restrict__ S,
                                                  const float* __restrict__ labels,
                                                  float* __restrict__ out) {
  int b = threadIdx.x;                   // 0..63
  float s0 = S[(size_t)b * BQ + b];                   // n=0 diagonal
  float s1 = S[(size_t)BQ * BQ + (size_t)b * BQ + b]; // n=1 diagonal

  // KL on paired scores
  float mx  = fmaxf(s0, s1);
  float lse = mx + logf(expf(s0 - mx) + expf(s1 - mx));
  float l0 = labels[b * NDOC + 0], l1 = labels[b * NDOC + 1];
  float kl = expf(l0) * (l0 - (s0 - lse)) + expf(l1) * (l1 - (s1 - lse));

  // in-batch CE over pred[b, n*B+bd] = S[n, b, bd], target column = b (n=0)
  float m2 = -INFINITY;
  for (int n = 0; n < NDOC; ++n)
    for (int d = 0; d < BQ; ++d)
      m2 = fmaxf(m2, S[(size_t)n * BQ * BQ + (size_t)b * BQ + d]);
  float se = 0.f;
  for (int n = 0; n < NDOC; ++n)
    for (int d = 0; d < BQ; ++d)
      se += expf(S[(size_t)n * BQ * BQ + (size_t)b * BQ + d] - m2);
  float ce = -(s0 - (m2 + logf(se)));

  float contrib = kl * (1.0f / BQ) + INBATCH_P * ce * (1.0f / BQ);

  __shared__ float red[64];
  red[b] = contrib;
  __syncthreads();
  for (int s = 32; s > 0; s >>= 1) {
    if (b < s) red[b] += red[b + s];
    __syncthreads();
  }
  if (b == 0) out[0] = red[0];
}

// ---------------------------------------------------------------------------
// Launch. Workspace layout: Qbf (512KB) | Dbf (8MB) | S (32KB)
// ---------------------------------------------------------------------------
extern "C" void kernel_launch(void* const* d_in, const int* in_sizes, int n_in,
                              void* d_out, int out_size, void* d_ws, size_t ws_size,
                              hipStream_t stream) {
  const float* q_reps  = (const float*)d_in[0];
  const float* d_reps  = (const float*)d_in[1];
  const int*   d_masks = (const int*)d_in[2];
  const float* labels  = (const float*)d_in[3];

  char* ws = (char*)d_ws;
  __bf16* Qbf = (__bf16*)ws;                                        // 64*32*128*2  = 512KB
  __bf16* Dbf = (__bf16*)(ws + 512u * 1024u);                       // 2*64*256*128*2 = 8MB
  float*  S   = (float*)(ws + 512u * 1024u + 8u * 1024u * 1024u);   // 2*64*64*4 = 32KB

  norm_q_kernel<<<dim3((BQ * LQ) / 8), 256, 0, stream>>>(q_reps, Qbf);
  norm_d_kernel<<<dim3(NDOC * BQ), 128, 0, stream>>>(d_reps, d_masks, Dbf);
  maxsim_kernel<<<dim3(NDOC * BQ, 4), 256, 0, stream>>>(Qbf, Dbf, S);
  loss_kernel<<<1, 64, 0, stream>>>(S, labels, (float*)d_out);
}
